// TransformerBlock_42606075576587
// MI455X (gfx1250) — matrix operand order repaired and checked by reading, not compiled, not run on hardware
//
#include <hip/hip_runtime.h>

typedef __attribute__((ext_vector_type(16))) __bf16 v16bf;
typedef __attribute__((ext_vector_type(8)))  float  v8f;

typedef unsigned short u16;
typedef unsigned int   u32;

union FragB16 {
    v16bf v;
    u16   h[16];
    uint4 q[2];
};

__device__ __forceinline__ u16 f2bf(float x) {
    u32 u = __builtin_bit_cast(u32, x);
    u32 r = u + 0x7fffu + ((u >> 16) & 1u);   // round-to-nearest-even
    return (u16)(r >> 16);
}

// CDNA5 async memory->LDS copy (one wave op = 32 lanes x 16B), ASYNCcnt-tracked.
__device__ __forceinline__ void async_copy_b128(u32 lds_off, const void* gaddr) {
    asm volatile("global_load_async_to_lds_b128 %0, %1, off"
                 :: "v"(lds_off), "v"(gaddr) : "memory");
}
#define WAIT_ASYNC(n) asm volatile("s_wait_asynccnt " #n ::: "memory")

// ---------------------------------------------------------------------------
// f32 -> bf16 elementwise convert
// ---------------------------------------------------------------------------
__global__ void cvt_f32_bf16(const float* __restrict__ src, u16* __restrict__ dst, int n) {
    int i = blockIdx.x * blockDim.x + threadIdx.x;
    if (i < n) dst[i] = f2bf(src[i]);
}

// ---------------------------------------------------------------------------
// W[K][N] f32  ->  Wt[N][K] bf16   (tiled transpose through LDS)
// ---------------------------------------------------------------------------
__global__ void transpose_cvt(const float* __restrict__ W, u16* __restrict__ Wt, int K, int N) {
    __shared__ float tile[32][33];
    int k0 = blockIdx.x * 32, n0 = blockIdx.y * 32;
    int tx = threadIdx.x, ty = threadIdx.y;           // block (32, 8)
    for (int i = ty; i < 32; i += 8)
        tile[i][tx] = W[(size_t)(k0 + i) * N + n0 + tx];
    __syncthreads();
    for (int i = ty; i < 32; i += 8)
        Wt[(size_t)(n0 + i) * K + k0 + tx] = f2bf(tile[tx][i]);
}

// ---------------------------------------------------------------------------
// Tiled bf16 WMMA GEMM: C = act(A[M,K] * Bt[N,K]^T + bias) * scale (+ resid)
// Block tile 128x128, BK=32 (one v_wmma_f32_16x16x32_bf16 K-step).
// 8 waves: 4 along M (32 rows) x 2 along N (64 cols) -> 8 accumulators/wave.
// Global->LDS staging via double-buffered GLOBAL_LOAD_ASYNC_TO_LDS_B128.
// K-loop is unrolled over the two buffers (K must be a multiple of 64) so
// buffer indices are compile-time constants -> no cndmask address selects.
// Optional outputs: f32, bf16 row-major, bf16 transposed (col-major, ld=M).
// ---------------------------------------------------------------------------
#define BM 128
#define BN 128
#define BK 32

__launch_bounds__(256)
__global__ void gemm_bf16(const u16* __restrict__ A, const u16* __restrict__ Bt,
                          const float* __restrict__ bias, const float* __restrict__ resid,
                          float* __restrict__ Cf, u16* __restrict__ Cb, u16* __restrict__ CbT,
                          int M, int N, int K, float scale, int act) {
    __shared__ __align__(16) u16 As[2][BM][BK];
    __shared__ __align__(16) u16 Bs[2][BN][BK];

    const int tid  = threadIdx.x;
    const int lane = tid & 31;
    const int wid  = tid >> 5;
    const int c    = lane & 15;
    const int hh   = lane >> 4;
    const int wm   = wid & 3;     // wave row  (32 M-rows)
    const int wn   = wid >> 2;    // wave col  (64 N-cols)
    const int m0   = blockIdx.y * BM;
    const int n0   = blockIdx.x * BN;

    const int ldRow = tid >> 1;          // 0..127
    const int ldSeg = (tid & 1) * 16;    // 0 or 16 (u16 elements)

    const u16* aPtr = A  + (size_t)(m0 + ldRow) * K + ldSeg;
    const u16* bPtr = Bt + (size_t)(n0 + ldRow) * K + ldSeg;

    v8f acc[2][4];
    for (int i = 0; i < 2; ++i)
        for (int j = 0; j < 4; ++j)
            for (int e = 0; e < 8; ++e) acc[i][j][e] = 0.f;

    auto issue = [&](const int bfr, int k0) {
        async_copy_b128((u32)(uintptr_t)&As[bfr][ldRow][ldSeg],     aPtr + k0);
        async_copy_b128((u32)(uintptr_t)&As[bfr][ldRow][ldSeg + 8], aPtr + k0 + 8);
        async_copy_b128((u32)(uintptr_t)&Bs[bfr][ldRow][ldSeg],     bPtr + k0);
        async_copy_b128((u32)(uintptr_t)&Bs[bfr][ldRow][ldSeg + 8], bPtr + k0 + 8);
    };

    auto compute = [&](const int bfr) {
        FragB16 af[2], bfg[4];
#pragma unroll
        for (int mt = 0; mt < 2; ++mt) {            // A lane layout: h[i]=K(8hh+i), h[8+i]=K(16+8hh+i)
            const u16* p = &As[bfr][wm * 32 + mt * 16 + c][hh * 8];
            af[mt].q[0] = *reinterpret_cast<const uint4*>(p);
            af[mt].q[1] = *reinterpret_cast<const uint4*>(p + 16);
        }
#pragma unroll
        for (int nt = 0; nt < 4; ++nt) {            // B lane layout: h[i]=K(8hh+i), h[8+i]=K(16+8hh+i), col=c
            const u16* p = &Bs[bfr][wn * 64 + nt * 16 + c][hh * 8];
            bfg[nt].q[0] = *reinterpret_cast<const uint4*>(p);
            bfg[nt].q[1] = *reinterpret_cast<const uint4*>(p + 16);
        }
#pragma unroll
        for (int mt = 0; mt < 2; ++mt)
#pragma unroll
            for (int nt = 0; nt < 4; ++nt)
                acc[mt][nt] = __builtin_amdgcn_wmma_f32_16x16x32_bf16(
                    false, af[mt].v, false, bfg[nt].v, (short)0, acc[mt][nt], false, false);
    };

    issue(0, 0);   // prefetch first tile into buffer 0
    for (int k0 = 0; k0 < K; k0 += 2 * BK) {
        // ---- buffer 0 ----
        issue(1, k0 + BK);      // always valid: K is a multiple of 2*BK
        WAIT_ASYNC(0x4);        // in-order completion: buffer 0's 4 copies done
        __syncthreads();
        compute(0);
        __syncthreads();        // readers done before buffer 0 is overwritten
        // ---- buffer 1 ----
        if (k0 + 2 * BK < K) {
            issue(0, k0 + 2 * BK);
            WAIT_ASYNC(0x4);
        } else {
            WAIT_ASYNC(0x0);
        }
        __syncthreads();
        compute(1);
        __syncthreads();
    }

    // Epilogue. C lane layout: acc[j] = D[row = j + 8*hh][col = c]
    for (int nt = 0; nt < 4; ++nt) {
        int ncol = n0 + wn * 64 + nt * 16 + c;
        float bv = bias ? bias[ncol] : 0.f;
        for (int mt = 0; mt < 2; ++mt) {
#pragma unroll
            for (int j = 0; j < 8; ++j) {
                int row = m0 + wm * 32 + mt * 16 + j + 8 * hh;
                float x = acc[mt][nt][j] + bv;
                if (act == 1) x = x / (1.f + __expf(-x));   // SiLU
                x *= scale;
                if (resid) x += resid[(size_t)row * N + ncol];
                if (Cf)  Cf[(size_t)row * N + ncol]  = x;
                if (Cb)  Cb[(size_t)row * N + ncol]  = f2bf(x);
                if (CbT) CbT[(size_t)ncol * M + row] = f2bf(x);
            }
        }
    }
}

// ---------------------------------------------------------------------------
// Causal flash attention. Q,K bf16 row-major [B*L, D] (Q pre-scaled 1/8),
// Vt bf16 col-major [D rows][B*L cols]. One wave = 16 q rows; computes
// S^T = K * Q^T so softmax stats are in-lane and exp(P) is already in the
// A-fragment layout for the P*V WMMA.
// ---------------------------------------------------------------------------
constexpr int AD  = 1024;  // model dim
constexpr int AH  = 16;    // heads
constexpr int AHD = 64;    // head dim
constexpr int AL  = 2048;  // sequence length

__launch_bounds__(256)
__global__ void attn_flash(const u16* __restrict__ Q, const u16* __restrict__ Kc,
                           const u16* __restrict__ Vt, u16* __restrict__ Y) {
    const int tid = threadIdx.x, lane = tid & 31, wid = tid >> 5;
    const int c = lane & 15, hh = lane >> 4;
    const int q0 = blockIdx.x * 128 + wid * 16;
    const int b  = blockIdx.y >> 4;
    const int h  = blockIdx.y & 15;

    // Q^T B-fragments (kstep 0: hd 0..31, kstep 1: hd 32..63), kept in VGPRs
    FragB16 qf[2];
    const size_t qbase = (size_t)(b * AL + q0 + c) * AD + h * AHD;
#pragma unroll
    for (int s = 0; s < 2; ++s) {
        const u16* p = Q + qbase + s * 32 + hh * 8;
        qf[s].q[0] = *reinterpret_cast<const uint4*>(p);
        qf[s].q[1] = *reinterpret_cast<const uint4*>(p + 16);
    }

    float m = -1e30f, l = 0.f;
    v8f o[4];
    for (int nt = 0; nt < 4; ++nt)
        for (int e = 0; e < 8; ++e) o[nt][e] = 0.f;

    for (int kv0 = 0; kv0 < q0 + 16; kv0 += 32) {
        // S^T tiles: M = 32 keys (two 16-key tiles), N = 16 q rows
        v8f st[2];
#pragma unroll
        for (int t = 0; t < 2; ++t) {
            v8f sa;
            for (int e = 0; e < 8; ++e) sa[e] = 0.f;
#pragma unroll
            for (int s = 0; s < 2; ++s) {
                FragB16 ka;  // A-fragment of K-tile: row = key (lane c), K-dim = hd
                const u16* p = Kc + (size_t)(b * AL + kv0 + 16 * t + c) * AD + h * AHD + s * 32 + hh * 8;
                ka.q[0] = *reinterpret_cast<const uint4*>(p);
                ka.q[1] = *reinterpret_cast<const uint4*>(p + 16);
                sa = __builtin_amdgcn_wmma_f32_16x16x32_bf16(
                    false, ka.v, false, qf[s].v, (short)0, sa, false, false);
            }
            st[t] = sa;
        }

        // causal mask + online softmax (lane c owns q-row q0+c)
        float pmax = -1e30f;
#pragma unroll
        for (int t = 0; t < 2; ++t)
#pragma unroll
            for (int j = 0; j < 8; ++j) {
                int key = kv0 + 16 * t + 8 * hh + j;
                if (key > q0 + c) st[t][j] = -1e30f;
                pmax = fmaxf(pmax, st[t][j]);
            }
        pmax = fmaxf(pmax, __shfl_xor(pmax, 16, 32));
        float mnew = fmaxf(m, pmax);
        float es   = __expf(m - mnew);
        m = mnew;

        float lsum = 0.f;
        FragB16 pf;  // exp(P) drops straight into the A-fragment layout
#pragma unroll
        for (int j = 0; j < 8; ++j) {
            float p0 = __expf(st[0][j] - m);
            float p1 = __expf(st[1][j] - m);
            lsum += p0 + p1;
            pf.h[j]     = f2bf(p0);
            pf.h[8 + j] = f2bf(p1);
        }
        lsum += __shfl_xor(lsum, 16, 32);
        l = l * es + lsum;

        float esj[8];  // broadcast per-row rescale to O's row mapping (row = j + 8*hh)
#pragma unroll
        for (int j = 0; j < 8; ++j) esj[j] = __shfl(es, j + 8 * hh + 16 * hh, 32);

#pragma unroll
        for (int nt = 0; nt < 4; ++nt) {
#pragma unroll
            for (int j = 0; j < 8; ++j) o[nt][j] *= esj[j];
            FragB16 vf;  // B-fragment of V: h[i] = V[key 8hh+i][hd nt*16+c], h[8+i] = V[key 16+8hh+i][hd nt*16+c]
            const u16* p = Vt + (size_t)(h * AHD + nt * 16 + c) * (2 * AL) + b * AL + kv0 + hh * 8;
            vf.q[0] = *reinterpret_cast<const uint4*>(p);
            vf.q[1] = *reinterpret_cast<const uint4*>(p + 16);
            o[nt] = __builtin_amdgcn_wmma_f32_16x16x32_bf16(
                false, pf.v, false, vf.v, (short)0, o[nt], false, false);
        }
    }

    float linv = 1.f / l;
    float lij[8];
#pragma unroll
    for (int j = 0; j < 8; ++j) lij[j] = __shfl(linv, j + 8 * hh + 16 * hh, 32);
    for (int nt = 0; nt < 4; ++nt)
#pragma unroll
        for (int j = 0; j < 8; ++j) {
            size_t idx = (size_t)(b * AL + q0 + j + 8 * hh) * AD + h * AHD + nt * 16 + c;
            Y[idx] = f2bf(o[nt][j] * lij[j]);
        }
}

// ---------------------------------------------------------------------------
extern "C" void kernel_launch(void* const* d_in, const int* in_sizes, int n_in,
                              void* d_out, int out_size, void* d_ws, size_t ws_size,
                              hipStream_t stream) {
    (void)in_sizes; (void)n_in; (void)out_size; (void)ws_size;
    const float* x  = (const float*)d_in[0];
    const float* Wq = (const float*)d_in[1];
    const float* bq = (const float*)d_in[2];
    const float* Wk = (const float*)d_in[3];
    const float* bk = (const float*)d_in[4];
    const float* Wv = (const float*)d_in[5];
    const float* bv = (const float*)d_in[6];
    const float* Wo = (const float*)d_in[7];
    const float* bo = (const float*)d_in[8];
    const float* W1 = (const float*)d_in[9];
    const float* b1 = (const float*)d_in[10];
    const float* W2 = (const float*)d_in[11];
    const float* b2 = (const float*)d_in[12];

    const int Dm = 1024, F = 4096, M = 4096;  // M = B*L

    char*  ws  = (char*)d_ws;
    size_t off = 0;
    auto alloc = [&](size_t bytes) -> void* {
        off = (off + 255) & ~(size_t)255;
        void* p = ws + off;
        off += bytes;
        return p;
    };

    u16* xb   = (u16*)alloc((size_t)M * Dm * 2);
    u16* wqT  = (u16*)alloc((size_t)Dm * Dm * 2);
    u16* wkT  = (u16*)alloc((size_t)Dm * Dm * 2);
    u16* wvT  = (u16*)alloc((size_t)Dm * Dm * 2);
    u16* woT  = (u16*)alloc((size_t)Dm * Dm * 2);
    u16* w1T  = (u16*)alloc((size_t)Dm * F * 2);
    u16* w2T  = (u16*)alloc((size_t)F * Dm * 2);
    u16* qb   = (u16*)alloc((size_t)M * Dm * 2);
    u16* kb   = (u16*)alloc((size_t)M * Dm * 2);
    u16* vtb  = (u16*)alloc((size_t)M * Dm * 2);   // V transposed: [Dm][M]
    u16* yb   = (u16*)alloc((size_t)M * Dm * 2);
    u16* x2b  = (u16*)alloc((size_t)M * Dm * 2);
    float* x2f = (float*)alloc((size_t)M * Dm * 4);
    u16* hb   = (u16*)alloc((size_t)M * F * 2);

    // 1. convert x to bf16
    {
        int n = M * Dm;
        cvt_f32_bf16<<<(n + 255) / 256, 256, 0, stream>>>(x, xb, n);
    }
    // 2. transpose-convert weights
    dim3 tb(32, 8);
    transpose_cvt<<<dim3(Dm / 32, Dm / 32), tb, 0, stream>>>(Wq, wqT, Dm, Dm);
    transpose_cvt<<<dim3(Dm / 32, Dm / 32), tb, 0, stream>>>(Wk, wkT, Dm, Dm);
    transpose_cvt<<<dim3(Dm / 32, Dm / 32), tb, 0, stream>>>(Wv, wvT, Dm, Dm);
    transpose_cvt<<<dim3(Dm / 32, Dm / 32), tb, 0, stream>>>(Wo, woT, Dm, Dm);
    transpose_cvt<<<dim3(Dm / 32, F / 32), tb, 0, stream>>>(W1, w1T, Dm, F);
    transpose_cvt<<<dim3(F / 32, Dm / 32), tb, 0, stream>>>(W2, w2T, F, Dm);

    // 3. Q/K/V projections (Q scaled by 1/sqrt(64); V stored transposed)
    dim3 gP(Dm / BN, M / BM);
    gemm_bf16<<<gP, 256, 0, stream>>>(xb, wqT, bq, nullptr, nullptr, qb, nullptr,
                                      M, Dm, Dm, 0.125f, 0);
    gemm_bf16<<<gP, 256, 0, stream>>>(xb, wkT, bk, nullptr, nullptr, kb, nullptr,
                                      M, Dm, Dm, 1.0f, 0);
    gemm_bf16<<<gP, 256, 0, stream>>>(xb, wvT, bv, nullptr, nullptr, nullptr, vtb,
                                      M, Dm, Dm, 1.0f, 0);

    // 4. causal flash attention -> Y (bf16 row-major)
    attn_flash<<<dim3(AL / 128, 2 * AH), 256, 0, stream>>>(qb, kb, vtb, yb);

    // 5. x2 = x + Y @ Wo + bo   (f32 + bf16 outputs)
    gemm_bf16<<<gP, 256, 0, stream>>>(yb, woT, bo, x, x2f, x2b, nullptr,
                                      M, Dm, Dm, 1.0f, 0);

    // 6. h = silu(x2 @ W1 + b1)  (bf16)
    gemm_bf16<<<dim3(F / BN, M / BM), 256, 0, stream>>>(x2b, w1T, b1, nullptr, nullptr, hb, nullptr,
                                                        M, F, Dm, 1.0f, 1);

    // 7. out = x2 + h @ W2 + b2  (f32 -> d_out)
    gemm_bf16<<<gP, 256, 0, stream>>>(hb, w2T, b2, x2f, (float*)d_out, nullptr, nullptr,
                                      M, Dm, F, 1.0f, 0);
}
